// Attention_16174846837077
// MI455X (gfx1250) — matrix operand order repaired and checked by reading, not compiled, not run on hardware
//
#include <hip/hip_runtime.h>
#include <hip/hip_bf16.h>

// ---------------------------------------------------------------------------
// MI455X / gfx1250 implementation: wave32, WMMA f32<-f16 16x16x32.
// Round 2: 2x2 register tiling in GEMMs (2x operand reuse per WMMA),
//          global_prefetch on streamed B operand.
// ---------------------------------------------------------------------------

typedef _Float16 f16;
typedef __attribute__((ext_vector_type(16))) _Float16 v16h;
typedef __attribute__((ext_vector_type(8)))  float    v8f;
typedef __attribute__((ext_vector_type(4)))  unsigned int u32x4;

#define NTOK   1024
#define CH     768
#define NHEAD  12
#define HDIM   64
#define BATCH  8
#define SCALE  0.125f            // 64^-0.5
#define OUT_ELEMS (BATCH*NTOK*CH)

// ---------------- WMMA fragment helpers (contiguous k order, same in A and B) --------------
// A 16x32 (MxK), f16: lanes 0-15 hold M=lane, K = k0..k0+15;
//                     lanes 16-31 hold M=lane-16, K = k0+16..k0+31.
__device__ __forceinline__ v16h load_a_frag(const f16* __restrict__ src, int stride,
                                            int m0, int k0, int lane) {
    int hs = lane >> 4;                       // half-select
    const f16* p0 = src + (size_t)(m0 + (lane & 15)) * stride + k0 + hs * 16;
    v16h a;
    *(u32x4*)&a       = *(const u32x4*)p0;        // elems 0..7
    *((u32x4*)&a + 1) = *(const u32x4*)(p0 + 8);  // elems 8..15
    return a;
}

// B 32x16 (KxN), f16 (stored so B[k][n] is contiguous in k for fixed n):
// lanes 0-15: N=lane, K=k0..k0+15 ; lanes 16-31: N=lane-16, K=k0+16..k0+31.
__device__ __forceinline__ v16h load_b_frag(const f16* __restrict__ src, int stride,
                                            int n0, int k0, int lane) {
    const f16* p = src + (size_t)(n0 + (lane & 15)) * stride + k0 + (lane >> 4) * 16;
    v16h b;
    *(u32x4*)&b       = *(const u32x4*)p;
    *((u32x4*)&b + 1) = *(const u32x4*)(p + 8);
    return b;
}

__device__ __forceinline__ v8f wmma_f16(v16h a, v16h b, v8f c) {
    return __builtin_amdgcn_wmma_f32_16x16x32_f16(false, a, false, b,
                                                  (short)0, c, false, false);
}

// C/D 16x16 f32: lanes 0-15 -> N=lane, VGPR v -> M=v ; lanes 16-31 -> N=lane-16, M=8+v.

// ---------------------------- fp32 -> f16 -----------------------------------
__global__ void cvt_f32_f16(const float* __restrict__ src, f16* __restrict__ dst, int n) {
    int i = blockIdx.x * blockDim.x + threadIdx.x;
    if (i < n) dst[i] = (f16)src[i];
}

// ----------------------- QKV C-tile scatter helper ---------------------------
__device__ __forceinline__ void store_qkv_tile(const v8f& acc, int n0, int mglob,
                                               int lane,
                                               f16* __restrict__ Qb,
                                               f16* __restrict__ Kb,
                                               f16* __restrict__ VT) {
    const int q3  = n0 / CH;
    const int rem = n0 % CH;
    const int h   = rem >> 6;
    const int dd0 = rem & 63;
    const int bi  = mglob >> 10;
    const int mrw = mglob & 1023;
    const int mloc = (lane >> 4) * 8;
    const int nloc = lane & 15;

    if (q3 == 2) {
        // transposed V: VT[(bh*64 + d)*1024 + token]; 8 consecutive tokens -> b128
        f16 tmp[8];
        #pragma unroll
        for (int v = 0; v < 8; ++v) tmp[v] = (f16)acc[v];
        f16* p = VT + ((size_t)(bi * NHEAD + h) * HDIM + dd0 + nloc) * NTOK + mrw + mloc;
        *(u32x4*)p = *(const u32x4*)tmp;
    } else {
        const float s = (q3 == 0) ? SCALE : 1.0f;   // fold softmax scale into Q
        f16* base = (q3 == 0 ? Qb : Kb)
                  + ((size_t)(bi * NHEAD + h) * NTOK + mrw + mloc) * HDIM + dd0 + nloc;
        #pragma unroll
        for (int v = 0; v < 8; ++v) base[(size_t)v * HDIM] = (f16)(acc[v] * s);
    }
}

// ------------------------------ QKV GEMM ------------------------------------
// M = 8192 tokens, N = 2304 (3*C), K = 768.  grid (256, 9), 256 threads (8 waves).
// Block stages 32 token rows in LDS (48KB); each wave computes a 32x32 output
// patch (2x2 WMMA tiles) -> each A/B fragment feeds 2 WMMAs.
__global__ __launch_bounds__(256)
void qkv_gemm(const f16* __restrict__ X, const f16* __restrict__ Wq,
              f16* __restrict__ Qb, f16* __restrict__ Kb, f16* __restrict__ VT) {
    __shared__ f16 xs[32 * CH];               // 48 KB: 32 token rows
    const int lane = threadIdx.x & 31;
    const int wave = threadIdx.x >> 5;
    const int m0   = blockIdx.x * 32;         // global token row base

    {   // cooperative contiguous copy (rows are contiguous in memory)
        const u32x4* s = (const u32x4*)(X + (size_t)m0 * CH);
        u32x4* d = (u32x4*)xs;
        for (int i = threadIdx.x; i < 32 * CH / 8; i += blockDim.x) d[i] = s[i];
    }
    __syncthreads();

    const int nt0 = blockIdx.y * 16 + wave * 2;   // first of 2 n-tiles (0..143)
    const int n0  = nt0 * 16;
    const f16* pB = Wq + (size_t)(n0 + (lane & 15)) * CH + (lane >> 4) * 16;

    v8f acc00 = {}, acc01 = {}, acc10 = {}, acc11 = {};
    for (int k0 = 0; k0 < CH; k0 += 32) {
        __builtin_prefetch(pB + k0 + 32, 0, 1);           // global_prefetch_b8
        v16h a0 = load_a_frag(xs, CH, 0,  k0, lane);
        v16h a1 = load_a_frag(xs, CH, 16, k0, lane);
        v16h b0 = load_b_frag(Wq, CH, n0,      k0, lane);
        v16h b1 = load_b_frag(Wq, CH, n0 + 16, k0, lane);
        acc00 = wmma_f16(a0, b0, acc00);
        acc01 = wmma_f16(a0, b1, acc01);
        acc10 = wmma_f16(a1, b0, acc10);
        acc11 = wmma_f16(a1, b1, acc11);
    }

    store_qkv_tile(acc00, n0,      m0,      lane, Qb, Kb, VT);
    store_qkv_tile(acc01, n0 + 16, m0,      lane, Qb, Kb, VT);
    store_qkv_tile(acc10, n0,      m0 + 16, lane, Qb, Kb, VT);
    store_qkv_tile(acc11, n0 + 16, m0 + 16, lane, Qb, Kb, VT);
}

// ------------------------------ Attention -----------------------------------
// grid (96, 32): one block per (b,h) x 32 query rows.  256 threads = 8 waves.
// Dynamic LDS: S 32x1024 f32 (128KB) + P 32x1024 f16 (64KB) = 192KB (<=320KB/WGP).
__global__ __launch_bounds__(256)
void attn(const f16* __restrict__ Qb, const f16* __restrict__ Kb,
          const f16* __restrict__ VT, f16* __restrict__ Ob) {
    extern __shared__ unsigned char smem[];
    float* S = (float*)smem;                       // [32][1024]
    f16*   P = (f16*)(smem + 32 * NTOK * 4);       // [32][1024]

    const int bh   = blockIdx.x;
    const int q0   = blockIdx.y * 32;
    const int lane = threadIdx.x & 31;
    const int wave = threadIdx.x >> 5;
    const f16* Qp = Qb + (size_t)bh * NTOK * HDIM;
    const f16* Kp = Kb + (size_t)bh * NTOK * HDIM;
    const f16* Vp = VT + (size_t)bh * HDIM * NTOK;

    // ---- pass A: S = (Q*scale) K^T  (scale folded into Q) ----
    const int rt = wave & 1;
    const int m0 = q0 + rt * 16;
    v16h qa0 = load_a_frag(Qp, HDIM, m0, 0,  lane);
    v16h qa1 = load_a_frag(Qp, HDIM, m0, 32, lane);
    #pragma unroll 4
    for (int j = 0; j < 16; ++j) {
        const int ct = (wave >> 1) + j * 4;        // key col tile 0..63
        v16h kb0 = load_b_frag(Kp, HDIM, ct * 16, 0,  lane);
        v16h kb1 = load_b_frag(Kp, HDIM, ct * 16, 32, lane);
        v8f s = {};
        s = wmma_f16(qa0, kb0, s);
        s = wmma_f16(qa1, kb1, s);
        const int col   = ct * 16 + (lane & 15);
        const int rbase = rt * 16 + (lane >> 4) * 8;
        #pragma unroll
        for (int v = 0; v < 8; ++v) S[(rbase + v) * NTOK + col] = s[v];
    }
    __syncthreads();

    // ---- pass B: row softmax (8 lanes per row, shfl_xor reductions, wave32) ----
    {
        const int r   = threadIdx.x >> 3;          // 0..31
        const int seg = threadIdx.x & 7;
        float* Sr = S + r * NTOK + seg * 128;
        f16*   Pr = P + r * NTOK + seg * 128;
        float mx = -1e30f;
        for (int i = 0; i < 128; ++i) mx = fmaxf(mx, Sr[i]);
        for (int o = 1; o < 8; o <<= 1) mx = fmaxf(mx, __shfl_xor(mx, o, 32));
        float sum = 0.f;
        for (int i = 0; i < 128; ++i) {
            float e = __expf(Sr[i] - mx);
            sum += e;
            Pr[i] = (f16)e;
        }
        for (int o = 1; o < 8; o <<= 1) sum += __shfl_xor(sum, o, 32);
        const float inv = 1.0f / sum;
        for (int i = 0; i < 128; ++i) Pr[i] = (f16)((float)Pr[i] * inv);
    }
    __syncthreads();

    // ---- pass C: O = P @ V  (A from LDS, B from transposed V: both contiguous) ----
    const int ct = wave >> 1;                      // output d-tile 0..3
    v8f acc = {};
    for (int k0 = 0; k0 < NTOK; k0 += 32) {
        v16h pa = load_a_frag(P,  NTOK, rt * 16, k0, lane);
        v16h vb = load_b_frag(Vp, NTOK, ct * 16, k0, lane);
        acc = wmma_f16(pa, vb, acc);
    }
    const int bi = bh / NHEAD, h = bh % NHEAD;
    const int nloc = lane & 15, mloc = (lane >> 4) * 8;
    f16* out = Ob + ((size_t)bi * NTOK + q0 + rt * 16 + mloc) * CH
                  + h * HDIM + ct * 16 + nloc;
    #pragma unroll
    for (int v = 0; v < 8; ++v) out[(size_t)v * CH] = (f16)acc[v];
}

// ---------------------------- Projection ------------------------------------
// out = O @ Wp^T + b  for batches 0-3; gumbel-combined K-range partials for 4-7.
// grid (256, 6), 256 threads.  Block stages 32 O-rows (48KB LDS); each wave
// computes 2 m-tiles for one n-tile (B fragments reused 2x), with 3 K-range
// accumulators per tile (T1: k<256, T2: 256<=k<384, T3: 384<=k<768).
__global__ __launch_bounds__(256)
void proj(const f16* __restrict__ Ob, const f16* __restrict__ Wp,
          const float* __restrict__ bias, const float* __restrict__ g,
          float* __restrict__ out) {
    __shared__ f16 os[32 * CH];
    const int lane = threadIdx.x & 31;
    const int wave = threadIdx.x >> 5;
    const int m0   = blockIdx.x * 32;

    {
        const u32x4* s = (const u32x4*)(Ob + (size_t)m0 * CH);
        u32x4* d = (u32x4*)os;
        for (int i = threadIdx.x; i < 32 * CH / 8; i += blockDim.x) d[i] = s[i];
    }
    __syncthreads();

    const int ntile = blockIdx.y * 8 + wave;       // 0..47
    const int n0 = ntile * 16;
    const f16* pB = Wp + (size_t)(n0 + (lane & 15)) * CH + (lane >> 4) * 16;

    v8f a1[2] = {{}, {}}, a2[2] = {{}, {}}, a3[2] = {{}, {}};
    for (int k0 = 0; k0 < 256; k0 += 32) {         // heads 0-3
        __builtin_prefetch(pB + k0 + 32, 0, 1);
        v16h b = load_b_frag(Wp, CH, n0, k0, lane);
        a1[0] = wmma_f16(load_a_frag(os, CH, 0,  k0, lane), b, a1[0]);
        a1[1] = wmma_f16(load_a_frag(os, CH, 16, k0, lane), b, a1[1]);
    }
    for (int k0 = 256; k0 < 384; k0 += 32) {       // heads 4-5
        __builtin_prefetch(pB + k0 + 32, 0, 1);
        v16h b = load_b_frag(Wp, CH, n0, k0, lane);
        a2[0] = wmma_f16(load_a_frag(os, CH, 0,  k0, lane), b, a2[0]);
        a2[1] = wmma_f16(load_a_frag(os, CH, 16, k0, lane), b, a2[1]);
    }
    for (int k0 = 384; k0 < CH; k0 += 32) {        // heads 6-11
        __builtin_prefetch(pB + k0 + 32, 0, 1);
        v16h b = load_b_frag(Wp, CH, n0, k0, lane);
        a3[0] = wmma_f16(load_a_frag(os, CH, 0,  k0, lane), b, a3[0]);
        a3[1] = wmma_f16(load_a_frag(os, CH, 16, k0, lane), b, a3[1]);
    }

    const int bi   = m0 >> 10;
    const int c    = n0 + (lane & 15);
    const int mloc = (lane >> 4) * 8;
    const float bv = bias[c];

    float w1, w2, w3, wb;
    if (bi < BATCH / 2) {                          // large branch: plain proj
        w1 = w2 = w3 = wb = 1.0f;
    } else {                                       // gumbel-gated small branch
        const float g0 = g[0], g1 = g[1], g2 = g[2];
        if (n0 < 256)      { w1 = g0 + g1 + g2; w2 = g0 + g1; w3 = g0; wb = w1; }
        else if (n0 < 384) { w1 = g0 + g1;      w2 = g0 + g1; w3 = g0; wb = w1; }
        else               { w1 = g0; w2 = g0; w3 = g0; wb = g0; }
    }

    #pragma unroll
    for (int mt = 0; mt < 2; ++mt) {
        float* dst = out + ((size_t)m0 + mt * 16 + mloc) * CH + c;
        #pragma unroll
        for (int v = 0; v < 8; ++v)
            dst[(size_t)v * CH] = w1 * a1[mt][v] + w2 * a2[mt][v]
                                + w3 * a3[mt][v] + wb * bv;
    }
}

// ------------------------------- Tail ---------------------------------------
__global__ void tail(const float* __restrict__ g, float* __restrict__ out) {
    if (threadIdx.x == 0) {
        float lat = 0.0f;
        #pragma unroll
        for (int i = 0; i < 3; ++i) {
            const long long a  = i + 1;
            const long long t1 = 3221225472LL / a;                 // (4*N*N*C//H)*H//a
            const long long t2 = ((1572864LL / a) * 768LL) / a;    // 2*N*C//a*C//a
            lat += (float)(t1 + t2) * g[i];
        }
        out[OUT_ELEMS + 0] = lat;
        out[OUT_ELEMS + 1] = g[0];
        out[OUT_ELEMS + 2] = g[1];
        out[OUT_ELEMS + 3] = g[2];
    }
}

// ----------------------------------------------------------------------------
static inline size_t align_up(size_t x) { return (x + 255) & ~size_t(255); }

extern "C" void kernel_launch(void* const* d_in, const int* in_sizes, int n_in,
                              void* d_out, int out_size, void* d_ws, size_t ws_size,
                              hipStream_t stream) {
    const float* x     = (const float*)d_in[0];   // (8,1024,768)
    const float* g     = (const float*)d_in[1];   // (3,)
    const float* wqkv  = (const float*)d_in[2];   // (2304,768)
    const float* wproj = (const float*)d_in[3];   // (768,768)
    const float* bproj = (const float*)d_in[4];   // (768,)
    float* out = (float*)d_out;

    const size_t NX  = (size_t)BATCH * NTOK * CH;     // 6291456
    const size_t NWQ = (size_t)3 * CH * CH;           // 1769472
    const size_t NWP = (size_t)CH * CH;               // 589824
    const size_t NQKV = (size_t)BATCH * NHEAD * NTOK * HDIM;  // == NX

    unsigned char* ws = (unsigned char*)d_ws;
    f16* xh  = (f16*)ws; ws += align_up(NX  * 2);
    f16* wqh = (f16*)ws; ws += align_up(NWQ * 2);
    f16* wph = (f16*)ws; ws += align_up(NWP * 2);
    f16* Qb  = (f16*)ws; ws += align_up(NQKV * 2);
    f16* Kb  = (f16*)ws; ws += align_up(NQKV * 2);
    f16* VT  = (f16*)ws; ws += align_up(NQKV * 2);
    f16* Ob  = (f16*)ws; ws += align_up(NX  * 2);

    cvt_f32_f16<<<dim3((NX  + 255) / 256), 256, 0, stream>>>(x,     xh,  (int)NX);
    cvt_f32_f16<<<dim3((NWQ + 255) / 256), 256, 0, stream>>>(wqkv,  wqh, (int)NWQ);
    cvt_f32_f16<<<dim3((NWP + 255) / 256), 256, 0, stream>>>(wproj, wph, (int)NWP);

    qkv_gemm<<<dim3(256, 9), 256, 0, stream>>>(xh, wqh, Qb, Kb, VT);

    const size_t attn_lds = 32 * NTOK * 4 + 32 * NTOK * 2;   // 192 KB (<=320KB/WGP)
    attn<<<dim3(BATCH * NHEAD, NTOK / 32), 256, attn_lds, stream>>>(Qb, Kb, VT, Ob);

    proj<<<dim3(256, 6), 256, 0, stream>>>(Ob, wph, bproj, g, out);
    tail<<<1, 32, 0, stream>>>(g, out);
}
